// BidirectionalRNN_11622181503290
// MI455X (gfx1250) — matrix operand order repaired and checked by reading, not compiled, not run on hardware
//
#include <hip/hip_runtime.h>

// ---------------- problem constants ----------------
#define BATCH 64
#define SEQ   512
#define DIN   300
#define HID   600
#define KD    320              // DIN padded to mult of 32
#define KH    640              // HID padded to mult of 32
#define NJ    608              // per-gate N padded to mult of 16
#define NG    (4*NJ)           // 2432 packed gate columns
#define KDT   (KD/32)          // 10
#define KHT   (KH/32)          // 20
#define NT    (NG/16)          // 152
#define JT    (NJ/16)          // 38
#define MBIG  (SEQ*BATCH)      // 32768 rows of the big GEMM (m = t*64 + b)
#define MBT   (MBIG/16)        // 2048 M-tiles

typedef __attribute__((ext_vector_type(16))) __bf16 v16bf;
typedef __attribute__((ext_vector_type(8)))  float  v8f;
typedef unsigned short u16t;
typedef unsigned int   u32t;

// ---------------- helpers ----------------
__device__ __forceinline__ u16t f2bf(float f) {
  u32t u = __float_as_uint(f);
  u = u + 0x7fffu + ((u >> 16) & 1u);      // round-to-nearest-even
  return (u16t)(u >> 16);
}

union BFV { uint4 q[2]; v16bf v; };

__device__ __forceinline__ v16bf ldv16(const u16t* p) {
  BFV u;
  const uint4* q = (const uint4*)p;        // 32B per lane -> 2x b128 loads
  u.q[0] = q[0];
  u.q[1] = q[1];
  return u.v;
}

__device__ __forceinline__ v8f wmma_bf16(v16bf a, v16bf b, v8f c) {
  return __builtin_amdgcn_wmma_f32_16x16x32_bf16(false, a, false, b, (short)0, c,
                                                 false, false);
}

// CDNA5 async Global->LDS copy (no VGPR round trip, tracked by ASYNCcnt).
__device__ __forceinline__ void async_copy_b128(unsigned lds_dst, const char* gsrc) {
  asm volatile("global_load_async_to_lds_b128 %0, %1, off"
               :: "v"(lds_dst), "v"(gsrc) : "memory");
}
__device__ __forceinline__ void wait_async0() {
  asm volatile("s_wait_asynccnt 0" ::: "memory");
}

// ---------------- kernels ----------------

// Zero h-state (double buffered), c-state, accumulators (contiguous region).
__global__ void k_zero(u32t* p, int n) {
  int i = blockIdx.x * blockDim.x + threadIdx.x;
  if (i < n) p[i] = 0u;
}

// Pack W (B-layout, K=320), U (B-layout, K=640), bias (padded) for both dirs.
__global__ void k_pack_wub(const float* __restrict__ Wf, const float* __restrict__ Uf,
                           const float* __restrict__ bf, const float* __restrict__ Wb,
                           const float* __restrict__ Ub, const float* __restrict__ bb,
                           u16t* __restrict__ wp, u16t* __restrict__ up,
                           float* __restrict__ bp) {
  const int WPE = 2 * NT * KDT * 512;
  const int UPE = 2 * NT * KHT * 512;
  const int BPE = 2 * NG;
  const int total = WPE + UPE + BPE;
  for (int idx = blockIdx.x * blockDim.x + threadIdx.x; idx < total;
       idx += gridDim.x * blockDim.x) {
    if (idx < WPE) {
      int i = idx;
      int dir = i / (NT * KDT * 512); i %= NT * KDT * 512;
      int nt = i / (KDT * 512);       i %= KDT * 512;
      int kt = i / 512;               i %= 512;
      int lane = i >> 4, e = i & 15;
      int k = kt * 32 + ((((e >> 3) << 4) | (e & 7)) | ((lane >> 4) << 3));  // B layout: table-order k
      int n = nt * 16 + (lane & 15);      // lane%16 = n col
      int gate = n / NJ, j = n % NJ;
      const float* W = dir ? Wb : Wf;
      float v = (k < DIN && j < HID) ? W[k * (4 * HID) + gate * HID + j] : 0.0f;
      wp[idx] = f2bf(v);
    } else if (idx < WPE + UPE) {
      int i = idx - WPE;
      int dir = i / (NT * KHT * 512); i %= NT * KHT * 512;
      int nt = i / (KHT * 512);       i %= KHT * 512;
      int kt = i / 512;               i %= 512;
      int lane = i >> 4, e = i & 15;
      int k = kt * 32 + ((((e >> 3) << 4) | (e & 7)) | ((lane >> 4) << 3));
      int n = nt * 16 + (lane & 15);
      int gate = n / NJ, j = n % NJ;
      const float* U = dir ? Ub : Uf;
      float v = (k < HID && j < HID) ? U[k * (4 * HID) + gate * HID + j] : 0.0f;
      up[idx - WPE] = f2bf(v);
    } else {
      int i = idx - WPE - UPE;
      int dir = i / NG;
      int n = i % NG;
      int gate = n / NJ, j = n % NJ;
      const float* bsrc = dir ? bb : bf;
      bp[i] = (j < HID) ? bsrc[gate * HID + j] : 0.0f;
    }
  }
}

// Pack inputs into WMMA A-layout (bf16), fwd and reverse-sequence variants.
// Row index m = t*64 + b.  A map: lane = ml + 16*((k>>3)&1), e = 8*(k>>4)+(k&7).
__global__ void k_pack_x(const float* __restrict__ x, const int* __restrict__ len,
                         u16t* __restrict__ xp) {
  const long long total = 2LL * MBT * KDT * 512;
  for (long long idx = blockIdx.x * (long long)blockDim.x + threadIdx.x; idx < total;
       idx += (long long)gridDim.x * blockDim.x) {
    long long i = idx;
    int dir = (int)(i / ((long long)MBT * KDT * 512)); i %= (long long)MBT * KDT * 512;
    int mt = (int)(i / (KDT * 512)); i %= KDT * 512;
    int kt = (int)(i / 512);         i %= 512;
    int lane = (int)(i >> 4), e = (int)(i & 15);
    int ml = lane & 15, hi = lane >> 4;
    int kl = ((e >> 3) << 4) | (e & 7) | (hi << 3);   // inverse A map
    int m = mt * 16 + ml;
    int t = m >> 6, b = m & 63;
    int k = kt * 32 + kl;
    float v = 0.0f;
    if (k < DIN) {
      int ts = t;
      if (dir) { int L = len[b]; ts = (t < L) ? (L - 1 - t) : t; }
      v = x[((long long)b * SEQ + ts) * DIN + k];
    }
    xp[idx] = f2bf(v);
  }
}

// xz is stored in C-fragment-packed layout: per 16x16 tile, 256 floats at
// lane*8 + v.  Tile index: ((dir*MBT + mt_big)*NT + nt).  Since m = t*64+b,
// mt_big = t*4 + batch_tile, so the step kernel reads its tile contiguously.
__device__ __forceinline__ void store_tile(float* __restrict__ xz,
                                           const float* __restrict__ bp, int dir,
                                           int mt, int nt, int lane, v8f a) {
  int nl = lane & 15;
  float bia = bp[dir * NG + nt * 16 + nl];
  float* p = xz + (((size_t)dir * MBT + mt) * NT + nt) * 256 + lane * 8;
  #pragma unroll
  for (int v = 0; v < 8; ++v) p[v] = a[v] + bia;    // 2x global_store_b128
}

// Big parallel GEMM: xz = xpack @ Wpack + bias, both directions.
// Each wave: 2x2 output tile block, K = 10 WMMA steps per tile.
__global__ void __launch_bounds__(256) k_xz_gemm(const u16t* __restrict__ xp,
                                                 const u16t* __restrict__ wp,
                                                 const float* __restrict__ bp,
                                                 float* __restrict__ xz) {
  const int BLK = (MBT / 2) * (NT / 2);   // 1024 * 76 wave-blocks per dir
  int wid = blockIdx.x * 8 + (threadIdx.x >> 5);
  int lane = threadIdx.x & 31;
  int dir = wid / BLK;
  int r = wid % BLK;
  int bm = r / (NT / 2);
  int bn = r % (NT / 2);
  const u16t* A = xp + (size_t)dir * MBT * KDT * 512;
  const u16t* B = wp + (size_t)dir * NT * KDT * 512;
  int mt0 = bm * 2, nt0 = bn * 2;
  v8f a00 = {}, a01 = {}, a10 = {}, a11 = {};
  for (int kt = 0; kt < KDT; ++kt) {
    v16bf A0 = ldv16(A + ((size_t)(mt0 * KDT + kt)) * 512 + lane * 16);
    v16bf A1 = ldv16(A + ((size_t)((mt0 + 1) * KDT + kt)) * 512 + lane * 16);
    v16bf B0 = ldv16(B + ((size_t)(nt0 * KDT + kt)) * 512 + lane * 16);
    v16bf B1 = ldv16(B + ((size_t)((nt0 + 1) * KDT + kt)) * 512 + lane * 16);
    a00 = wmma_bf16(A0, B0, a00);
    a01 = wmma_bf16(A0, B1, a01);
    a10 = wmma_bf16(A1, B0, a10);
    a11 = wmma_bf16(A1, B1, a11);
  }
  store_tile(xz, bp, dir, mt0,     nt0,     lane, a00);
  store_tile(xz, bp, dir, mt0,     nt0 + 1, lane, a01);
  store_tile(xz, bp, dir, mt0 + 1, nt0,     lane, a10);
  store_tile(xz, bp, dir, mt0 + 1, nt0 + 1, lane, a11);
}

// One recurrence step, both directions (blockIdx.y), 38 hidden-unit tiles
// (blockIdx.x). Each WG computes all 4 gates for its 16 hidden units so the
// c/h update is local. The per-WG working set (80KB h A-tiles + 80KB U
// B-tiles) is staged in LDS via async global->LDS copies; WMMA operands are
// then fed from LDS (ds_load_b128). h-state is double buffered across step
// launches to avoid cross-WG races.
__global__ void __launch_bounds__(256) k_step(int t, const float* __restrict__ xz,
                                              const u16t* __restrict__ up,
                                              u16t* __restrict__ hp,
                                              float* __restrict__ cs,
                                              float* __restrict__ acc,
                                              const int* __restrict__ len) {
  __shared__ u16t sH[4 * KHT * 512];      // 80 KB: A tiles (all mt, all kt)
  __shared__ u16t sU[4 * KHT * 512];      // 80 KB: B tiles, 4 gates at this jt
  __shared__ float zl[4 * BATCH * 16];    // 16 KB: [gate][b][nl]
  int dir = blockIdx.y;
  int jt  = blockIdx.x;
  int w = threadIdx.x >> 5, lane = threadIdx.x & 31;
  int mt = w >> 1;                        // batch tile 0..3
  int g0 = (w & 1) * 2;                   // gates {g0, g0+1}
  const u16t* Hr = hp + (size_t)(dir * 2 + (t & 1)) * 4 * KHT * 512;
  u16t*       Hw = hp + (size_t)(dir * 2 + ((t + 1) & 1)) * 4 * KHT * 512;
  const u16t* U  = up + (size_t)dir * NT * KHT * 512;

  // ---- async fill of LDS (160 KB total, 640 B per thread) ----
  {
    unsigned ldsH = (unsigned)(size_t)&sH[0];
    const char* gH = (const char*)Hr;
    #pragma unroll
    for (int it = 0; it < 20; ++it) {     // 81920 B / (256 thr * 16 B)
      int off = (threadIdx.x + it * 256) * 16;
      async_copy_b128(ldsH + off, gH + off);
    }
    unsigned ldsU = (unsigned)(size_t)&sU[0];
    #pragma unroll
    for (int g = 0; g < 4; ++g) {         // 4 chunks of 20480 B (contiguous)
      const char* gU = (const char*)(U + (size_t)((g * JT + jt) * KHT) * 512);
      #pragma unroll
      for (int it = 0; it < 5; ++it) {
        int off = (threadIdx.x + it * 256) * 16;
        async_copy_b128(ldsU + g * 20480 + off, gU + off);
      }
    }
    wait_async0();
    __syncthreads();
  }

  // ---- GEMM: z = h @ U for 2 tiles per wave (shared A operand) ----
  v8f c0 = {}, c1 = {};
  for (int kt = 0; kt < KHT; ++kt) {
    v16bf Av = ldv16(&sH[(mt * KHT + kt) * 512 + lane * 16]);
    v16bf B0 = ldv16(&sU[(g0 * KHT + kt) * 512 + lane * 16]);
    v16bf B1 = ldv16(&sU[((g0 + 1) * KHT + kt) * 512 + lane * 16]);
    c0 = wmma_bf16(Av, B0, c0);
    c1 = wmma_bf16(Av, B1, c1);
  }

  // ---- add xz (C-packed, contiguous per lane) and stage z in LDS ----
  int nl = lane & 15;
  int mbase = (lane >> 4) * 8;
  int mt_big = t * 4 + mt;
  const float* XZ0 =
      xz + (((size_t)dir * MBT + mt_big) * NT + (g0 * JT + jt)) * 256 + lane * 8;
  const float* XZ1 =
      xz + (((size_t)dir * MBT + mt_big) * NT + ((g0 + 1) * JT + jt)) * 256 + lane * 8;
  #pragma unroll
  for (int v = 0; v < 8; ++v) {
    int b = mt * 16 + mbase + v;
    zl[g0 * (BATCH * 16) + b * 16 + nl]       = c0[v] + XZ0[v];
    zl[(g0 + 1) * (BATCH * 16) + b * 16 + nl] = c1[v] + XZ1[v];
  }
  __syncthreads();

  // ---- gates, state update, masked accumulation, repack h ----
  float* Cs = cs + (size_t)dir * BATCH * NJ;
  float* Ac = acc + (size_t)dir * BATCH * NJ;
  #pragma unroll
  for (int r = 0; r < 4; ++r) {
    int idx = r * 256 + threadIdx.x;      // (b, jl) pairs, 1024 total
    int b = idx >> 4, jl = idx & 15;
    int j = jt * 16 + jl;
    float zi = zl[0 * (BATCH * 16) + b * 16 + jl];
    float zf = zl[1 * (BATCH * 16) + b * 16 + jl];
    float zg = zl[2 * (BATCH * 16) + b * 16 + jl];
    float zo = zl[3 * (BATCH * 16) + b * 16 + jl];
    float iv = 1.0f / (1.0f + __expf(-zi));
    float fv = 1.0f / (1.0f + __expf(-zf));
    float gv = tanhf(zg);
    float ov = 1.0f / (1.0f + __expf(-zo));
    float c_new = fv * Cs[b * NJ + j] + iv * gv;
    float h_new = ov * tanhf(c_new);
    if (j >= HID) { c_new = 0.0f; h_new = 0.0f; }   // pad units stay zero
    Cs[b * NJ + j] = c_new;
    if (t < len[b]) Ac[b * NJ + j] += h_new;        // masked running sum
    // scatter h into A-packed layout for next step's WMMA
    int mt2 = b >> 4, ml = b & 15;
    int kt2 = j >> 5, kl = j & 31;
    int lane2 = ml + 16 * ((kl >> 3) & 1);
    int e2 = ((kl >> 4) << 3) | (kl & 7);
    Hw[(size_t)(mt2 * KHT + kt2) * 512 + lane2 * 16 + e2] = f2bf(h_new);
  }
}

// out[b, h] = sum / T
__global__ void k_final(const float* __restrict__ acc, float* __restrict__ out) {
  int i = blockIdx.x * blockDim.x + threadIdx.x;
  if (i >= BATCH * 2 * HID) return;
  int b = i / (2 * HID), h = i % (2 * HID);
  float v = (h < HID) ? acc[b * NJ + h]
                      : acc[(size_t)BATCH * NJ + b * NJ + (h - HID)];
  out[i] = v * (1.0f / (float)SEQ);
}

// ---------------- launcher ----------------
extern "C" void kernel_launch(void* const* d_in, const int* in_sizes, int n_in,
                              void* d_out, int out_size, void* d_ws, size_t ws_size,
                              hipStream_t stream) {
  (void)in_sizes; (void)n_in; (void)out_size; (void)ws_size;
  const float* x   = (const float*)d_in[0];
  const int*   len = (const int*)d_in[1];
  // d_in[2] = training flag (unused, inference path)
  const float* Wf = (const float*)d_in[3];
  const float* Uf = (const float*)d_in[4];
  const float* bf = (const float*)d_in[5];
  const float* Wb = (const float*)d_in[6];
  const float* Ub = (const float*)d_in[7];
  const float* bb = (const float*)d_in[8];
  float* out = (float*)d_out;

  char* ws = (char*)d_ws;
  size_t off = 0;
  float* xz  = (float*)(ws + off); off += (size_t)2 * MBT * NT * 256 * sizeof(float);
  u16t*  xp  = (u16t*)(ws + off);  off += (size_t)2 * MBT * KDT * 512 * sizeof(u16t);
  u16t*  wpk = (u16t*)(ws + off);  off += (size_t)2 * NT * KDT * 512 * sizeof(u16t);
  u16t*  upk = (u16t*)(ws + off);  off += (size_t)2 * NT * KHT * 512 * sizeof(u16t);
  float* bpk = (float*)(ws + off); off += (size_t)2 * NG * sizeof(float);
  u16t*  hpk = (u16t*)(ws + off);  off += (size_t)2 * 2 * 4 * KHT * 512 * sizeof(u16t);
  float* cst = (float*)(ws + off); off += (size_t)2 * BATCH * NJ * sizeof(float);
  float* acc = (float*)(ws + off); off += (size_t)2 * BATCH * NJ * sizeof(float);

  // zero h-state (both buffers) + c-state + accumulators: contiguous region
  int zero_words = (int)(((size_t)2 * 2 * 4 * KHT * 512 * sizeof(u16t) +
                          (size_t)4 * BATCH * NJ * sizeof(float)) / 4);
  k_zero<<<(zero_words + 255) / 256, 256, 0, stream>>>((u32t*)hpk, zero_words);

  k_pack_wub<<<4096, 256, 0, stream>>>(Wf, Uf, bf, Wb, Ub, bb, wpk, upk, bpk);
  k_pack_x<<<8192, 256, 0, stream>>>(x, len, xp);

  // 2 dirs * 1024 * 76 wave-blocks, 8 waves per WG
  k_xz_gemm<<<(2 * (MBT / 2) * (NT / 2)) / 8, 256, 0, stream>>>(xp, wpk, bpk, xz);

  for (int t = 0; t < SEQ; ++t)
    k_step<<<dim3(JT, 2), 256, 0, stream>>>(t, xz, upk, hpk, cst, acc, len);

  k_final<<<(BATCH * 2 * HID + 255) / 256, 256, 0, stream>>>(acc, out);
}
